// ModelC_90417651515513
// MI455X (gfx1250) — matrix operand order repaired and checked by reading, not compiled, not run on hardware
//
#include <hip/hip_runtime.h>
#include <hip/hip_bf16.h>

// ---------------------------------------------------------------------------
// MI455X / gfx1250 implementation: bf16 WMMA (f32 accumulate) for all GEMMs
// and both attention matmuls; flash-style online softmax; deterministic
// segment means.  B=16, L=1024, D=256, H=4, HD=64, DFF=2048, LAT=3.
//
// GEMM blocking: one wave owns a 32x64 output strip (2 M-tiles x 4 N-tiles,
// 8 accumulators): 8 WMMAs per 12 b128 loads instead of 1 per 4, which keeps
// the WMMA pipe fed from L2 (everything here fits in the 192 MB L2).
// ---------------------------------------------------------------------------

typedef __bf16 bf16;
typedef __attribute__((ext_vector_type(16))) __bf16 v16bf;
typedef __attribute__((ext_vector_type(8)))  __bf16 v8bf;
typedef __attribute__((ext_vector_type(8)))  float  v8f;

union Frag { v16bf v; v8bf h[2]; };

__device__ __forceinline__ v8f wmma_bf16(const v16bf& a, const v16bf& b, v8f c) {
  // (neg_a, A, neg_b, B, c_mod, C, reuse_a, reuse_b)
  return __builtin_amdgcn_wmma_f32_16x16x32_bf16(false, a, false, b, (short)0, c,
                                                 false, false);
}

#define NB   16
#define LL   1024
#define DD   256
#define HH   4
#define HD   64
#define DFFV 2048
#define NR   (NB * LL)          // 16384 rows

// ---------------------------------------------------------------------------
// Elementwise helpers
// ---------------------------------------------------------------------------
__global__ void zero_f32(float* p, size_t n) {
  size_t i = (size_t)blockIdx.x * blockDim.x + threadIdx.x;
  if (i < n) p[i] = 0.f;
}

__global__ void cvt_bf16(const float* in, bf16* out, size_t n) {
  size_t i = (size_t)blockIdx.x * blockDim.x + threadIdx.x;
  if (i < n) out[i] = (bf16)in[i];
}

__global__ void embed_kernel(const int* si, const int* ci, const float* se,
                             const float* ce, float* xf, bf16* xb) {
  size_t i = (size_t)blockIdx.x * blockDim.x + threadIdx.x;   // N*D
  if (i >= (size_t)NR * DD) return;
  int d = (int)(i & (DD - 1));
  size_t row = i >> 8;
  float v = se[(size_t)si[row] * DD + d] + ce[(size_t)ci[row] * DD + d];
  xf[i] = v;
  xb[i] = (bf16)v;
}

// vT[b,h,hd,l] = qkv[(b*L+l)*768 + 512 + h*64 + hd]
__global__ void transpose_v(const bf16* qkv, bf16* vT) {
  size_t i = (size_t)blockIdx.x * blockDim.x + threadIdx.x;   // B*H*HD*L
  if (i >= (size_t)NB * HH * HD * LL) return;
  int l  = (int)(i & (LL - 1));
  size_t j = i >> 10;
  int hd = (int)(j & (HD - 1)); j >>= 6;
  int h  = (int)(j & (HH - 1));
  int b  = (int)(j >> 2);
  vT[i] = qkv[((size_t)(b * LL + l)) * (3 * DD) + 2 * DD + h * HD + hd];
}

// ---------------------------------------------------------------------------
// WMMA GEMM:  C[M,N] = A[M,K] * W[N,K]^T + bias, optional relu.
// One wave per 32x64 output strip (2x4 16x16 tiles); 4 waves per block.
// Requires M % 32 == 0, Nout % 64 == 0, K % 32 == 0 (true for all calls).
// ---------------------------------------------------------------------------
__global__ __launch_bounds__(128) void gemm_bf16_kernel(
    const bf16* __restrict__ A, int lda,
    const bf16* __restrict__ W, int ldw,
    const float* __restrict__ bias,
    float* __restrict__ Cf, bf16* __restrict__ Cb, int ldc,
    int M, int Nout, int K, int relu) {
  const int wave = threadIdx.x >> 5;
  const int lane = threadIdx.x & 31;
  const int stripsN = Nout >> 6;                 // 64-wide strips
  const int strip = blockIdx.x * 4 + wave;
  const int totStrips = (M >> 5) * stripsN;
  if (strip >= totStrips) return;                // wave-uniform exit
  const int sm = strip / stripsN;                // 32-row block index
  const int sn = strip - sm * stripsN;           // 64-col strip index
  const int l15 = lane & 15;
  const int hf  = lane >> 4;

  const bf16* arow0 = A + (size_t)(sm * 32 + l15) * lda;      // M-tile 0
  const bf16* arow1 = arow0 + (size_t)16 * lda;               // M-tile 1
  const bf16* wrow  = W + (size_t)(sn * 64 + l15) * ldw;      // N-tile 0 base
  const int kbA = hf * 16;       // A-frag: run  [kb,kb+16)
  const int koB = hf * 16;       // B-frag: run  [ko,ko+16)

  v8f acc[2][4] = {};
  for (int k0 = 0; k0 < K; k0 += 32) {
    Frag a0, a1;
    a0.h[0] = *(const v8bf*)(arow0 + k0 + kbA);
    a0.h[1] = *(const v8bf*)(arow0 + k0 + kbA + 8);
    a1.h[0] = *(const v8bf*)(arow1 + k0 + kbA);
    a1.h[1] = *(const v8bf*)(arow1 + k0 + kbA + 8);
    if (k0 + 32 < K) {                           // gfx1250 global_prefetch_b8
      __builtin_prefetch(arow0 + k0 + 64, 0, 1);
      __builtin_prefetch(wrow + k0 + 64, 0, 1);
    }
#pragma unroll
    for (int t = 0; t < 4; t++) {
      Frag b;
      const bf16* wr = wrow + (size_t)(t * 16) * ldw + k0 + koB;
      b.h[0] = *(const v8bf*)(wr);
      b.h[1] = *(const v8bf*)(wr + 8);
      acc[0][t] = wmma_bf16(a0.v, b.v, acc[0][t]);
      acc[1][t] = wmma_bf16(a1.v, b.v, acc[1][t]);
    }
  }

#pragma unroll
  for (int t = 0; t < 4; t++) {
    const int col = sn * 64 + t * 16 + l15;
    const float bz = bias ? bias[col] : 0.f;
#pragma unroll
    for (int mt = 0; mt < 2; mt++) {
#pragma unroll
      for (int r = 0; r < 8; r++) {
        const int rowg = sm * 32 + mt * 16 + hf * 8 + r;
        float v = acc[mt][t][r] + bz;
        if (relu) v = fmaxf(v, 0.f);
        if (Cf) Cf[(size_t)rowg * ldc + col] = v;
        if (Cb) Cb[(size_t)rowg * ldc + col] = (bf16)v;
      }
    }
  }
}

// ---------------------------------------------------------------------------
// Flash attention: one wave per (b, h, 16-row q-tile). K/V walked in 32-key
// steps; scores via 4 WMMAs, P staged through LDS (C-layout -> A-layout),
// ctx via 4 WMMAs against pre-transposed V^T.
// ---------------------------------------------------------------------------
__global__ __launch_bounds__(32) void attn_kernel(
    const bf16* __restrict__ qkv, const bf16* __restrict__ vT,
    bf16* __restrict__ ctxb) {
  const int lane = threadIdx.x;
  const int bid  = blockIdx.x;
  const int qt = bid & 63;             // L/16
  const int h  = (bid >> 6) & (HH - 1);
  const int b  = bid >> 8;
  const int row0 = b * LL + qt * 16;
  const int l15 = lane & 15;
  const int hf  = lane >> 4;

  const bf16* Q  = qkv + (size_t)row0 * (3 * DD) + h * HD;
  const bf16* Kp = qkv + (size_t)b * LL * (3 * DD) + DD + h * HD;
  const bf16* Vt = vT + ((size_t)(b * HH + h) * HD) * LL;   // [hd][L]

  v8f c0 = {}, c1 = {}, c2 = {}, c3 = {};
  float mrow[8], lrow[8];
#pragma unroll
  for (int r = 0; r < 8; r++) { mrow[r] = -1e30f; lrow[r] = 0.f; }

  // Preload Q A-fragments for hd-steps 0 and 32.
  Frag qa0, qa1;
  {
    const bf16* qrow = Q + (size_t)l15 * (3 * DD);
    const int kb = hf * 16;
    qa0.h[0] = *(const v8bf*)(qrow + kb);
    qa0.h[1] = *(const v8bf*)(qrow + kb + 8);
    qa1.h[0] = *(const v8bf*)(qrow + 32 + kb);
    qa1.h[1] = *(const v8bf*)(qrow + 32 + kb + 8);
  }

  __shared__ __align__(16) bf16 P[16 * 32];

  for (int k0 = 0; k0 < LL; k0 += 32) {
    // ---- scores: 16 q-rows x 32 keys, K-dim = HD = 64 ----
    v8f s0 = {}, s1 = {};
    {
      const bf16* kr0 = Kp + (size_t)(k0 + l15) * (3 * DD);
      const bf16* kr1 = Kp + (size_t)(k0 + 16 + l15) * (3 * DD);
      const int ko = hf * 16;
      Frag b00, b01, b10, b11;
      b00.h[0] = *(const v8bf*)(kr0 + ko);      b00.h[1] = *(const v8bf*)(kr0 + ko + 8);
      b01.h[0] = *(const v8bf*)(kr0 + 32 + ko); b01.h[1] = *(const v8bf*)(kr0 + 32 + ko + 8);
      b10.h[0] = *(const v8bf*)(kr1 + ko);      b10.h[1] = *(const v8bf*)(kr1 + ko + 8);
      b11.h[0] = *(const v8bf*)(kr1 + 32 + ko); b11.h[1] = *(const v8bf*)(kr1 + 32 + ko + 8);
      s0 = wmma_bf16(qa0.v, b00.v, s0);
      s0 = wmma_bf16(qa1.v, b01.v, s0);
      s1 = wmma_bf16(qa0.v, b10.v, s1);
      s1 = wmma_bf16(qa1.v, b11.v, s1);
    }

    // ---- online softmax (row-wise over 32 cols; rows live per (r, half)) ----
    float p0[8], p1[8];
#pragma unroll
    for (int r = 0; r < 8; r++) {
      float a0 = s0[r] * 0.125f;       // 1/sqrt(64)
      float a1 = s1[r] * 0.125f;
      float tmax = fmaxf(a0, a1);
#pragma unroll
      for (int o = 8; o > 0; o >>= 1)
        tmax = fmaxf(tmax, __shfl_xor(tmax, o, 32));
      const float mn = fmaxf(mrow[r], tmax);
      const float al = __expf(mrow[r] - mn);
      const float e0 = __expf(a0 - mn);
      const float e1 = __expf(a1 - mn);
      float sm = e0 + e1;
#pragma unroll
      for (int o = 8; o > 0; o >>= 1)
        sm += __shfl_xor(sm, o, 32);
      lrow[r] = lrow[r] * al + sm;
      mrow[r] = mn;
      p0[r] = e0; p1[r] = e1;
      c0[r] *= al; c1[r] *= al; c2[r] *= al; c3[r] *= al;
    }

    // ---- stage P (C-layout) into LDS, reload as A-fragment ----
#pragma unroll
    for (int r = 0; r < 8; r++) {
      P[(hf * 8 + r) * 32 + l15]      = (bf16)p0[r];
      P[(hf * 8 + r) * 32 + 16 + l15] = (bf16)p1[r];
    }
    __syncthreads();
    Frag pa;
    {
      const int kb = hf * 16;
      pa.h[0] = *(const v8bf*)(P + l15 * 32 + kb);
      pa.h[1] = *(const v8bf*)(P + l15 * 32 + kb + 8);
    }
    __syncthreads();

    // ---- ctx += P(16x32) * V(32x64): B-frags contiguous from V^T ----
    {
      const size_t kof = (size_t)k0 + hf * 16;
      Frag vb;
      const bf16* v0 = Vt + (size_t)(0 * 16 + l15) * LL + kof;
      vb.h[0] = *(const v8bf*)(v0); vb.h[1] = *(const v8bf*)(v0 + 8);
      c0 = wmma_bf16(pa.v, vb.v, c0);
      const bf16* v1 = Vt + (size_t)(1 * 16 + l15) * LL + kof;
      vb.h[0] = *(const v8bf*)(v1); vb.h[1] = *(const v8bf*)(v1 + 8);
      c1 = wmma_bf16(pa.v, vb.v, c1);
      const bf16* v2 = Vt + (size_t)(2 * 16 + l15) * LL + kof;
      vb.h[0] = *(const v8bf*)(v2); vb.h[1] = *(const v8bf*)(v2 + 8);
      c2 = wmma_bf16(pa.v, vb.v, c2);
      const bf16* v3 = Vt + (size_t)(3 * 16 + l15) * LL + kof;
      vb.h[0] = *(const v8bf*)(v3); vb.h[1] = *(const v8bf*)(v3 + 8);
      c3 = wmma_bf16(pa.v, vb.v, c3);
    }
  }

  // ---- epilogue: divide by row sum, store ctx (bf16, [N, D]) ----
#pragma unroll
  for (int r = 0; r < 8; r++) {
    const size_t grow = (size_t)(row0 + hf * 8 + r);
    const float inv = 1.f / lrow[r];
    ctxb[grow * DD + h * HD + 0 * 16 + l15] = (bf16)(c0[r] * inv);
    ctxb[grow * DD + h * HD + 1 * 16 + l15] = (bf16)(c1[r] * inv);
    ctxb[grow * DD + h * HD + 2 * 16 + l15] = (bf16)(c2[r] * inv);
    ctxb[grow * DD + h * HD + 3 * 16 + l15] = (bf16)(c3[r] * inv);
  }
}

// ---------------------------------------------------------------------------
// Residual + LayerNorm over D=256 (one 256-thread block per row).
// ---------------------------------------------------------------------------
__global__ __launch_bounds__(256) void add_ln_kernel(
    const float* __restrict__ a, const float* __restrict__ b,
    const float* __restrict__ g, const float* __restrict__ beta,
    float* __restrict__ of, bf16* __restrict__ ob) {
  const int row = blockIdx.x;
  const int t = threadIdx.x;
  const size_t idx = (size_t)row * DD + t;
  float v = a[idx] + b[idx];

  __shared__ float red[8];
  float s = v;
#pragma unroll
  for (int o = 16; o > 0; o >>= 1) s += __shfl_down(s, o, 32);
  if ((t & 31) == 0) red[t >> 5] = s;
  __syncthreads();
  float tot = 0.f;
#pragma unroll
  for (int i = 0; i < 8; i++) tot += red[i];
  const float mu = tot * (1.f / DD);
  __syncthreads();

  const float d = v - mu;
  float s2 = d * d;
#pragma unroll
  for (int o = 16; o > 0; o >>= 1) s2 += __shfl_down(s2, o, 32);
  if ((t & 31) == 0) red[t >> 5] = s2;
  __syncthreads();
  float tot2 = 0.f;
#pragma unroll
  for (int i = 0; i < 8; i++) tot2 += red[i];
  const float var = tot2 * (1.f / DD);
  const float y = d * rsqrtf(var + 1e-5f) * g[t] + beta[t];
  if (of) of[idx] = y;
  if (ob) ob[idx] = (bf16)y;
}

// ---------------------------------------------------------------------------
// Boundary detection: latent logits (LAT=3) + argmax==0; force l==0 -> false.
// ---------------------------------------------------------------------------
__global__ void boundary_kernel(const float* __restrict__ to,
                                const float* __restrict__ lw,
                                const float* __restrict__ lb,
                                int* __restrict__ close) {
  const int row = blockIdx.x * blockDim.x + threadIdx.x;
  if (row >= NR) return;
  float z0 = lb[0], z1 = lb[1], z2 = lb[2];
  const float* x = to + (size_t)row * DD;
  for (int d = 0; d < DD; d++) {
    const float xv = x[d];
    z0 += xv * lw[d];
    z1 += xv * lw[DD + d];
    z2 += xv * lw[2 * DD + d];
  }
  int arg = 0; float best = z0;
  if (z1 > best) { best = z1; arg = 1; }
  if (z2 > best) { best = z2; arg = 2; }
  int cl = (arg == 0) ? 1 : 0;
  if ((row & (LL - 1)) == 0) cl = 0;   // boundaries[:,0] never closes
  close[row] = cl;
}

// Exclusive cumsum of close per batch (serial per batch, 16 lanes).
__global__ void seg_kernel(const int* __restrict__ close, int* __restrict__ segid,
                           int* __restrict__ nseg) {
  const int b = threadIdx.x;
  if (b >= NB) return;
  int run = 0;
  for (int l = 0; l < LL; l++) {
    const int i = b * LL + l;
    segid[i] = run;
    run += close[i];
  }
  nseg[b] = run + 1;
}

// Find contiguous segment ranges (deterministic, no atomics).
__global__ void segrange_kernel(const int* __restrict__ close,
                                const int* __restrict__ segid,
                                int* __restrict__ segstart,
                                float* __restrict__ segcnt) {
  const int row = blockIdx.x * blockDim.x + threadIdx.x;
  if (row >= NR) return;
  const int l = row & (LL - 1);
  const int b = row >> 10;
  const bool start = (l == 0) || (close[row - 1] == 1);
  if (!start) return;
  int len = 1;
  while (((row + len) & (LL - 1)) != 0 && close[row + len - 1] == 0) len++;
  const int gid = b * LL + segid[row];
  segstart[gid] = row;
  segcnt[gid] = (float)len;
}

// Segment means -> bf16 (zero rows for unused segment slots).
__global__ void means_kernel(const float* __restrict__ to,
                             const int* __restrict__ segstart,
                             const float* __restrict__ segcnt,
                             bf16* __restrict__ mb) {
  size_t i = (size_t)blockIdx.x * blockDim.x + threadIdx.x;   // N*D
  if (i >= (size_t)NR * DD) return;
  const int d = (int)(i & (DD - 1));
  const size_t g = i >> 8;
  const float c = segcnt[g];
  float m = 0.f;
  if (c > 0.f) {
    const int r0 = segstart[g];
    float s = 0.f;
    const int len = (int)c;
    for (int j = 0; j < len; j++) s += to[(size_t)(r0 + j) * DD + d];
    m = s / c;
  }
  mb[i] = (bf16)m;
}

// Predicate head: p = sigmoid(h . w2 + b2), masked by segment validity.
__global__ void pred_kernel(const float* __restrict__ hp,
                            const float* __restrict__ w2,
                            const float* __restrict__ b2,
                            const float* __restrict__ segcnt,
                            float* __restrict__ pv) {
  const int row = blockIdx.x * blockDim.x + threadIdx.x;
  if (row >= NR) return;
  float acc = b2[0];
  const float* h = hp + (size_t)row * DD;
  for (int d = 0; d < DD; d++) acc += h[d] * w2[d];
  const float p = 1.f / (1.f + __expf(-acc));
  pv[row] = (segcnt[row] > 0.f) ? p : 0.f;
}

// Final per-batch reduction: out[b] = sum(p*valid) / n_seg.
__global__ __launch_bounds__(256) void out_kernel(const float* __restrict__ pv,
                                                  const int* __restrict__ nseg,
                                                  float* __restrict__ out) {
  const int b = blockIdx.x;
  const int t = threadIdx.x;
  float s = 0.f;
  for (int l = t; l < LL; l += 256) s += pv[b * LL + l];
  __shared__ float red[8];
#pragma unroll
  for (int o = 16; o > 0; o >>= 1) s += __shfl_down(s, o, 32);
  if ((t & 31) == 0) red[t >> 5] = s;
  __syncthreads();
  if (t == 0) {
    float tot = 0.f;
    for (int i = 0; i < 8; i++) tot += red[i];
    out[b] = tot / (float)nseg[b];
  }
}

// ---------------------------------------------------------------------------
// Host orchestration
// ---------------------------------------------------------------------------
static inline int cdiv(long a, long b) { return (int)((a + b - 1) / b); }

extern "C" void kernel_launch(void* const* d_in, const int* in_sizes, int n_in,
                              void* d_out, int out_size, void* d_ws, size_t ws_size,
                              hipStream_t stream) {
  (void)in_sizes; (void)n_in; (void)out_size; (void)ws_size;

  const int*   shape_idxs = (const int*)  d_in[0];
  const int*   color_idxs = (const int*)  d_in[1];
  const float* shape_emb  = (const float*)d_in[2];
  const float* color_emb  = (const float*)d_in[3];
  const float* in_proj_w  = (const float*)d_in[4];
  const float* in_proj_b  = (const float*)d_in[5];
  const float* out_proj_w = (const float*)d_in[6];
  const float* out_proj_b = (const float*)d_in[7];
  const float* ln1_g      = (const float*)d_in[8];
  const float* ln1_b      = (const float*)d_in[9];
  const float* ff_w1      = (const float*)d_in[10];
  const float* ff_b1      = (const float*)d_in[11];
  const float* ff_w2      = (const float*)d_in[12];
  const float* ff_b2      = (const float*)d_in[13];
  const float* ln2_g      = (const float*)d_in[14];
  const float* ln2_b      = (const float*)d_in[15];
  const float* latent_w   = (const float*)d_in[16];
  const float* latent_b   = (const float*)d_in[17];
  const float* pred_w1    = (const float*)d_in[18];
  const float* pred_b1    = (const float*)d_in[19];
  const float* pred_w2    = (const float*)d_in[20];
  const float* pred_b2    = (const float*)d_in[21];
  float* out = (float*)d_out;

  const size_t N = NR;
  size_t off = 0;
  auto arena = [&](size_t bytes) -> void* {
    void* r = (char*)d_ws + off;
    off += (bytes + 255) & ~(size_t)255;
    return r;
  };

  bf16* wb_qkv   = (bf16*)arena((size_t)3 * DD * DD * 2);
  bf16* wb_out   = (bf16*)arena((size_t)DD * DD * 2);
  bf16* wb_ff1   = (bf16*)arena((size_t)DFFV * DD * 2);
  bf16* wb_ff2   = (bf16*)arena((size_t)DD * DFFV * 2);
  bf16* wb_pred1 = (bf16*)arena((size_t)DD * DD * 2);

  float* xf    = (float*)arena(N * DD * 4);
  bf16*  xb    = (bf16*) arena(N * DD * 2);
  bf16*  qkvb  = (bf16*) arena(N * 3 * DD * 2);
  bf16*  vT    = (bf16*) arena(N * DD * 2);
  bf16*  ctxb  = (bf16*) arena(N * DD * 2);
  float* saf   = (float*)arena(N * DD * 4);
  float* x1f   = (float*)arena(N * DD * 4);
  bf16*  x1b   = (bf16*) arena(N * DD * 2);
  bf16*  hb    = (bf16*) arena(N * DFFV * 2);
  float* fff   = (float*)arena(N * DD * 4);
  float* tof   = (float*)arena(N * DD * 4);
  int*   close = (int*)  arena(N * 4);
  int*   segid = (int*)  arena(N * 4);
  int*   nseg  = (int*)  arena(64);
  int*   segst = (int*)  arena(N * 4);
  float* segcn = (float*)arena(N * 4);
  bf16*  meanb = (bf16*) arena(N * DD * 2);
  float* hpred = (float*)arena(N * DD * 4);
  float* pv    = (float*)arena(N * 4);

  // --- weight conversion fp32 -> bf16 ---
  cvt_bf16<<<cdiv((long)3 * DD * DD, 256), 256, 0, stream>>>(in_proj_w, wb_qkv, (size_t)3 * DD * DD);
  cvt_bf16<<<cdiv((long)DD * DD, 256), 256, 0, stream>>>(out_proj_w, wb_out, (size_t)DD * DD);
  cvt_bf16<<<cdiv((long)DFFV * DD, 256), 256, 0, stream>>>(ff_w1, wb_ff1, (size_t)DFFV * DD);
  cvt_bf16<<<cdiv((long)DD * DFFV, 256), 256, 0, stream>>>(ff_w2, wb_ff2, (size_t)DD * DFFV);
  cvt_bf16<<<cdiv((long)DD * DD, 256), 256, 0, stream>>>(pred_w1, wb_pred1, (size_t)DD * DD);

  // --- embeddings ---
  embed_kernel<<<cdiv((long)N * DD, 256), 256, 0, stream>>>(
      shape_idxs, color_idxs, shape_emb, color_emb, xf, xb);

  // --- QKV projection: [N,256] x [768,256]^T -> bf16 [N,768] ---
  {
    int strips = (NR / 32) * (3 * DD / 64);
    gemm_bf16_kernel<<<cdiv(strips, 4), 128, 0, stream>>>(
        xb, DD, wb_qkv, DD, in_proj_b, nullptr, qkvb, 3 * DD, NR, 3 * DD, DD, 0);
  }

  // --- V transpose for attention B-fragments ---
  transpose_v<<<cdiv((long)N * DD, 256), 256, 0, stream>>>(qkvb, vT);

  // --- attention ---
  attn_kernel<<<NB * HH * (LL / 16), 32, 0, stream>>>(qkvb, vT, ctxb);

  // --- out_proj: ctx [N,256] x [256,256]^T -> fp32 sa ---
  {
    int strips = (NR / 32) * (DD / 64);
    gemm_bf16_kernel<<<cdiv(strips, 4), 128, 0, stream>>>(
        ctxb, DD, wb_out, DD, out_proj_b, saf, nullptr, DD, NR, DD, DD, 0);
  }

  // --- LN1(x + sa) ---
  add_ln_kernel<<<NR, 256, 0, stream>>>(xf, saf, ln1_g, ln1_b, x1f, x1b);

  // --- FFN1 (relu): [N,256] x [2048,256]^T -> bf16 h ---
  {
    int strips = (NR / 32) * (DFFV / 64);
    gemm_bf16_kernel<<<cdiv(strips, 4), 128, 0, stream>>>(
        x1b, DD, wb_ff1, DD, ff_b1, nullptr, hb, DFFV, NR, DFFV, DD, 1);
  }

  // --- FFN2: [N,2048] x [256,2048]^T -> fp32 ff ---
  {
    int strips = (NR / 32) * (DD / 64);
    gemm_bf16_kernel<<<cdiv(strips, 4), 128, 0, stream>>>(
        hb, DFFV, wb_ff2, DFFV, ff_b2, fff, nullptr, DD, NR, DD, DFFV, 0);
  }

  // --- LN2(x1 + ff) -> trans_out ---
  add_ln_kernel<<<NR, 256, 0, stream>>>(x1f, fff, ln2_g, ln2_b, tof, nullptr);

  // --- boundaries / segments ---
  boundary_kernel<<<cdiv(NR, 256), 256, 0, stream>>>(tof, latent_w, latent_b, close);
  seg_kernel<<<1, 32, 0, stream>>>(close, segid, nseg);
  zero_f32<<<cdiv((long)NR, 256), 256, 0, stream>>>(segcn, NR);
  segrange_kernel<<<cdiv(NR, 256), 256, 0, stream>>>(close, segid, segst, segcn);
  means_kernel<<<cdiv((long)N * DD, 256), 256, 0, stream>>>(tof, segst, segcn, meanb);

  // --- predicate MLP layer 1 (relu) via WMMA ---
  {
    int strips = (NR / 32) * (DD / 64);
    gemm_bf16_kernel<<<cdiv(strips, 4), 128, 0, stream>>>(
        meanb, DD, wb_pred1, DD, pred_b1, hpred, nullptr, DD, NR, DD, DD, 1);
  }

  // --- predicate head + final reduction ---
  pred_kernel<<<cdiv(NR, 256), 256, 0, stream>>>(hpred, pred_w2, pred_b2, segcn, pv);
  out_kernel<<<NB, 256, 0, stream>>>(pv, nseg, out);
}
